// CrossAttention_11373073400358
// MI455X (gfx1250) — matrix operand order repaired and checked by reading, not compiled, not run on hardware
//
#include <hip/hip_runtime.h>
#include <hip/hip_bf16.h>

// ---------------------------------------------------------------------------
// CDNA5 / gfx1250 cross-attention:
//   - all GEMM stages via v_wmma_f32_16x16x32_bf16
//   - attention K tiles staged LDS via global_load_async_to_lds_b128
//     (double-buffered, s_wait_asynccnt + workgroup barrier)
// ---------------------------------------------------------------------------

typedef __attribute__((ext_vector_type(16))) __bf16 v16bf;
typedef __attribute__((ext_vector_type(8)))  __bf16 v8bf;
typedef __attribute__((ext_vector_type(8)))  float  v8f;

#define WMMA_BF16(a, b, c) \
    __builtin_amdgcn_wmma_f32_16x16x32_bf16(false, (a), false, (b), (short)0, (c), false, false)

union FragU { v16bf v; v8bf h[2]; };

// ---- A-fragment (16x32, bf16, contiguous-k order) ------------------------------
// lane L: row = rowBase + (L&15); K-halves: lanes 0-15 -> K 0..15,
// lanes 16-31 -> K 16..31.
__device__ __forceinline__ v16bf load_a_f32(const float* __restrict__ A, int rowBase,
                                            int ldk, int k0, int lane) {
    int row = rowBase + (lane & 15);
    int kk  = k0 + ((lane >> 4) << 4);
    const float* p = A + (size_t)row * ldk + kk;
    FragU u;
#pragma unroll
    for (int i = 0; i < 8; ++i) u.h[0][i] = (__bf16)p[i];
#pragma unroll
    for (int i = 0; i < 8; ++i) u.h[1][i] = (__bf16)p[8 + i];
    return u.v;
}

__device__ __forceinline__ v16bf load_a_bf16(const __bf16* __restrict__ A, int rowBase,
                                             int ldk, int k0, int lane) {
    int row = rowBase + (lane & 15);
    int kk  = k0 + ((lane >> 4) << 4);
    const __bf16* p = A + (size_t)row * ldk + kk;
    FragU u;
    u.h[0] = *(const v8bf*)p;
    u.h[1] = *(const v8bf*)(p + 8);
    return u.v;
}

// ---- B-fragment (32x16, bf16) ---------------------------------------------
// Source stored row-major as (N x K): lane L holds column n = colBase+(L&15),
// K-range (L>>4)*16 .. +15 -> one contiguous 32B load.
__device__ __forceinline__ v16bf load_b_f32(const float* __restrict__ W, int colBase,
                                            int ldk, int k0, int lane) {
    int col = colBase + (lane & 15);
    int kk  = k0 + ((lane >> 4) << 4);
    const float* p = W + (size_t)col * ldk + kk;
    v16bf r;
#pragma unroll
    for (int i = 0; i < 16; ++i) r[i] = (__bf16)p[i];
    return r;
}

__device__ __forceinline__ v16bf load_b_bf16(const __bf16* __restrict__ W, int colBase,
                                             int ldk, int k0, int lane) {
    int col = colBase + (lane & 15);
    int kk  = k0 + ((lane >> 4) << 4);
    return *(const v16bf*)(W + (size_t)col * ldk + kk);
}

// B-fragment from an LDS-resident 32x64 bf16 tile (key-major, 64-half rows).
__device__ __forceinline__ v16bf load_b_lds(const __bf16* buf, int keyBase,
                                            int k0, int lane) {
    int key = keyBase + (lane & 15);
    int kk  = k0 + ((lane >> 4) << 4);
    return *(const v16bf*)(buf + key * 64 + kk);
}

// ===========================================================================
// Kernel 1: Q = hidden @ Wq^T   (M=8192, K=1024, N=1024) -> bf16
// ===========================================================================
__global__ void __launch_bounds__(256) ca_proj_q(const float* __restrict__ Hin,
                                                 const float* __restrict__ Wq,
                                                 __bf16* __restrict__ Qb) {
    const int K = 1024, N = 1024;
    int lane = threadIdx.x & 31;
    int wid  = threadIdx.x >> 5;
    int rowBase = blockIdx.x * 64 + (wid >> 1) * 16;
    int colBase = blockIdx.y * 128 + (wid & 1) * 64;

    v8f acc[4];
#pragma unroll
    for (int t = 0; t < 4; ++t)
#pragma unroll
        for (int j = 0; j < 8; ++j) acc[t][j] = 0.f;

    for (int k0 = 0; k0 < K; k0 += 32) {
        if (k0 + 64 < K)   // pull the next A tile row toward the WGP
            __builtin_prefetch(Hin + (size_t)(rowBase + (lane & 15)) * K + k0 + 64, 0, 1);
        v16bf a = load_a_f32(Hin, rowBase, K, k0, lane);
#pragma unroll
        for (int t = 0; t < 4; ++t) {
            v16bf b = load_b_f32(Wq, colBase + t * 16, K, k0, lane);
            acc[t] = WMMA_BF16(a, b, acc[t]);
        }
    }
    int hi = lane >> 4, c0 = lane & 15;
#pragma unroll
    for (int t = 0; t < 4; ++t)
#pragma unroll
        for (int j = 0; j < 8; ++j) {
            int r = rowBase + j + hi * 8;
            int c = colBase + t * 16 + c0;
            Qb[(size_t)r * N + c] = (__bf16)acc[t][j];
        }
}

// ===========================================================================
// Kernel 2: KV = context @ Wkv^T  (M=2048, K=768, N=2048)
// cols 0..1023 -> K  (row-major keys x (h*64+d))
// cols 1024..2047 -> V^T  ((b,h,d) x keys)  for contiguous PV B-operand loads
// ===========================================================================
__global__ void __launch_bounds__(256) ca_proj_kv(const float* __restrict__ Ctx,
                                                  const float* __restrict__ Wkv,
                                                  __bf16* __restrict__ Kb,
                                                  __bf16* __restrict__ Vt) {
    const int K = 768;
    int lane = threadIdx.x & 31;
    int wid  = threadIdx.x >> 5;
    int rowBase = blockIdx.x * 64 + (wid >> 1) * 16;
    int colBase = blockIdx.y * 128 + (wid & 1) * 64;

    v8f acc[4];
#pragma unroll
    for (int t = 0; t < 4; ++t)
#pragma unroll
        for (int j = 0; j < 8; ++j) acc[t][j] = 0.f;

    for (int k0 = 0; k0 < K; k0 += 32) {
        if (k0 + 64 < K)
            __builtin_prefetch(Ctx + (size_t)(rowBase + (lane & 15)) * K + k0 + 64, 0, 1);
        v16bf a = load_a_f32(Ctx, rowBase, K, k0, lane);
#pragma unroll
        for (int t = 0; t < 4; ++t) {
            v16bf b = load_b_f32(Wkv, colBase + t * 16, K, k0, lane);
            acc[t] = WMMA_BF16(a, b, acc[t]);
        }
    }
    int hi = lane >> 4, c0 = lane & 15;
#pragma unroll
    for (int t = 0; t < 4; ++t)
#pragma unroll
        for (int j = 0; j < 8; ++j) {
            int r = rowBase + j + hi * 8;          // r = b*1024 + m
            int c = colBase + t * 16 + c0;
            __bf16 val = (__bf16)acc[t][j];
            if (c < 1024) {
                Kb[(size_t)r * 1024 + c] = val;
            } else {
                int cc = c - 1024;
                int h = cc >> 6, d = cc & 63;
                int b = r >> 10, m = r & 1023;
                Vt[((size_t)((b * 16 + h) * 64 + d)) * 1024 + m] = val;
            }
        }
}

// ===========================================================================
// Kernel 3: flash attention. Block = 128 thr = 4 waves; wave owns 16 q-rows.
// K chunk (32 keys x 64 dh, 4KB bf16) staged to LDS with
// global_load_async_to_lds_b128, double-buffered; synced via
// s_wait_asynccnt + workgroup barrier.  Per chunk: 4 score WMMAs, online
// softmax (xor-shuffle reductions in 16-lane C-layout groups), P C-layout ->
// LDS -> A-layout, 4 PV WMMAs (V^T rows key-contiguous from global).
// ===========================================================================

// each of the 128 threads copies 2 x 16B of the 4KB K chunk
__device__ __forceinline__ void stage_k_async(const __bf16* __restrict__ Kh, int kc,
                                              __bf16* dst, int tid) {
#pragma unroll
    for (int u = 0; u < 2; ++u) {
        int ch  = tid * 2 + u;              // 0..255 : 16B chunk index
        int key = ch >> 3;                  // 8 chunks per 128B key row
        int off = (ch & 7) * 8;             // half offset within row
        const __bf16* g = Kh + (size_t)(kc + key) * 1024 + off;
        unsigned           l  = (unsigned)(size_t)(dst + key * 64 + off);
        unsigned long long ga = (unsigned long long)(size_t)g;
        asm volatile("global_load_async_to_lds_b128 %0, %1, off"
                     :: "v"(l), "v"(ga) : "memory");
    }
}

__global__ void __launch_bounds__(128) ca_attention(const __bf16* __restrict__ Qb,
                                                    const __bf16* __restrict__ Kb,
                                                    const __bf16* __restrict__ Vt,
                                                    __bf16* __restrict__ Ob) {
    const int Nq = 4096, Nc = 1024;
    __shared__ __align__(32) __bf16 kbuf[2][32 * 64];   // double-buffered K tile
    __shared__ __align__(32) __bf16 pls[4 * 16 * 40];   // P transpose, per wave

    int idx  = blockIdx.x;
    int qblk = idx & 63;          // Nq/64 tiles
    int bh   = idx >> 6;
    int b    = bh >> 4;
    int h    = bh & 15;

    int tid  = threadIdx.x;
    int lane = tid & 31;
    int wid  = tid >> 5;
    int rowBase = qblk * 64 + wid * 16;
    int hi = lane >> 4, c0 = lane & 15;

    const __bf16* Qh = Qb + (size_t)b * Nq * 1024 + h * 64;       // row-stride 1024
    const __bf16* Kh = Kb + (size_t)b * Nc * 1024 + h * 64;       // row-stride 1024
    const __bf16* Vh = Vt + (size_t)((b * 16 + h) * 64) * 1024;   // d-major, key contiguous

    v16bf qf0 = load_a_bf16(Qh, rowBase, 1024, 0, lane);
    v16bf qf1 = load_a_bf16(Qh, rowBase, 1024, 32, lane);

    float mrow[8], lrow[8];
    v8f acc[4];
#pragma unroll
    for (int j = 0; j < 8; ++j) { mrow[j] = -1e30f; lrow[j] = 0.f; }
#pragma unroll
    for (int t = 0; t < 4; ++t)
#pragma unroll
        for (int j = 0; j < 8; ++j) acc[t][j] = 0.f;

    __bf16* pw = pls + wid * 640;

    // prime buffer 0
    stage_k_async(Kh, 0, kbuf[0], tid);

    for (int kc = 0; kc < Nc; kc += 32) {
        int it = kc >> 5;
        // current buffer finished loading (per-wave ASYNCcnt), then make it
        // visible block-wide
        asm volatile("s_wait_asynccnt 0x0" ::: "memory");
        __syncthreads();
        if (kc + 32 < Nc)
            stage_k_async(Kh, kc + 32, kbuf[(it + 1) & 1], tid);
        const __bf16* kb = kbuf[it & 1];

        // ---- scores: S(16x32) = Q(16x64) . K^T, two 16x16 C-tiles from LDS
        v8f s0, s1;
#pragma unroll
        for (int j = 0; j < 8; ++j) { s0[j] = 0.f; s1[j] = 0.f; }
        {
            v16bf bk = load_b_lds(kb, 0, 0, lane);
            s0 = WMMA_BF16(qf0, bk, s0);
            bk = load_b_lds(kb, 0, 32, lane);
            s0 = WMMA_BF16(qf1, bk, s0);
            bk = load_b_lds(kb, 16, 0, lane);
            s1 = WMMA_BF16(qf0, bk, s1);
            bk = load_b_lds(kb, 16, 32, lane);
            s1 = WMMA_BF16(qf1, bk, s1);
        }
        const float scale = 0.125f;   // 1/sqrt(64)
        float cmax[8];
#pragma unroll
        for (int j = 0; j < 8; ++j) {
            s0[j] *= scale; s1[j] *= scale;
            cmax[j] = fmaxf(s0[j], s1[j]);
        }
#pragma unroll
        for (int mask = 1; mask < 16; mask <<= 1)
#pragma unroll
            for (int j = 0; j < 8; ++j)
                cmax[j] = fmaxf(cmax[j], __shfl_xor(cmax[j], mask, 32));

        float esc[8], p0[8], p1[8], rs[8];
#pragma unroll
        for (int j = 0; j < 8; ++j) {
            float mn = fmaxf(mrow[j], cmax[j]);
            esc[j] = __expf(mrow[j] - mn);
            mrow[j] = mn;
            p0[j] = __expf(s0[j] - mn);
            p1[j] = __expf(s1[j] - mn);
            rs[j] = p0[j] + p1[j];
        }
#pragma unroll
        for (int mask = 1; mask < 16; mask <<= 1)
#pragma unroll
            for (int j = 0; j < 8; ++j)
                rs[j] += __shfl_xor(rs[j], mask, 32);
#pragma unroll
        for (int j = 0; j < 8; ++j) lrow[j] = lrow[j] * esc[j] + rs[j];
#pragma unroll
        for (int t = 0; t < 4; ++t)
#pragma unroll
            for (int j = 0; j < 8; ++j) acc[t][j] *= esc[j];

        // ---- P: C-layout -> LDS -> A-layout (per-wave tile, wave-local order)
#pragma unroll
        for (int j = 0; j < 8; ++j) {
            int r = j + hi * 8;
            pw[r * 40 + c0]      = (__bf16)p0[j];
            pw[r * 40 + 16 + c0] = (__bf16)p1[j];
        }
        asm volatile("s_wait_dscnt 0" ::: "memory");
        FragU up;
        {
            int rA = lane & 15;
            int kh = (lane >> 4) * 16;
            up.h[0] = *(const v8bf*)(pw + rA * 40 + kh);
            up.h[1] = *(const v8bf*)(pw + rA * 40 + kh + 8);
        }
        v16bf pA = up.v;

        // ---- O += P(16x32) . V(32x64); V^T rows are key-contiguous
#pragma unroll
        for (int t = 0; t < 4; ++t) {
            v16bf bv = *(const v16bf*)(Vh + (size_t)(t * 16 + c0) * 1024 + kc + hi * 16);
            acc[t] = WMMA_BF16(pA, bv, acc[t]);
        }
    }

    float inv[8];
#pragma unroll
    for (int j = 0; j < 8; ++j) inv[j] = 1.f / lrow[j];

    __bf16* Oh = Ob + (size_t)b * Nq * 1024 + h * 64;
#pragma unroll
    for (int t = 0; t < 4; ++t)
#pragma unroll
        for (int j = 0; j < 8; ++j)
            Oh[(size_t)(rowBase + j + hi * 8) * 1024 + t * 16 + c0] =
                (__bf16)(acc[t][j] * inv[j]);
}

// ===========================================================================
// Kernel 4: out = attn @ Wout^T + bout  (M=8192, K=1024, N=1024) -> f32
// ===========================================================================
__global__ void __launch_bounds__(256) ca_proj_out(const __bf16* __restrict__ Ain,
                                                   const float* __restrict__ Wout,
                                                   const float* __restrict__ bout,
                                                   float* __restrict__ Out) {
    const int K = 1024, N = 1024;
    int lane = threadIdx.x & 31;
    int wid  = threadIdx.x >> 5;
    int rowBase = blockIdx.x * 64 + (wid >> 1) * 16;
    int colBase = blockIdx.y * 128 + (wid & 1) * 64;

    v8f acc[4];
#pragma unroll
    for (int t = 0; t < 4; ++t)
#pragma unroll
        for (int j = 0; j < 8; ++j) acc[t][j] = 0.f;

    for (int k0 = 0; k0 < K; k0 += 32) {
        if (k0 + 64 < K)
            __builtin_prefetch(Ain + (size_t)(rowBase + (lane & 15)) * K + k0 + 64, 0, 1);
        v16bf a = load_a_bf16(Ain, rowBase, K, k0, lane);
#pragma unroll
        for (int t = 0; t < 4; ++t) {
            v16bf b = load_b_f32(Wout, colBase + t * 16, K, k0, lane);
            acc[t] = WMMA_BF16(a, b, acc[t]);
        }
    }
    int hi = lane >> 4, c0 = lane & 15;
#pragma unroll
    for (int t = 0; t < 4; ++t)
#pragma unroll
        for (int j = 0; j < 8; ++j) {
            int r = rowBase + j + hi * 8;
            int c = colBase + t * 16 + c0;
            Out[(size_t)r * N + c] = acc[t][j] + bout[c];
        }
}

// ===========================================================================
// Launcher
// ===========================================================================
extern "C" void kernel_launch(void* const* d_in, const int* in_sizes, int n_in,
                              void* d_out, int out_size, void* d_ws, size_t ws_size,
                              hipStream_t stream) {
    const float* hs   = (const float*)d_in[0];  // (2, 4096, 1024)
    const float* ctx  = (const float*)d_in[1];  // (2, 1024, 768)
    const float* Wq   = (const float*)d_in[2];  // (1024, 1024)
    const float* Wkv  = (const float*)d_in[3];  // (2048, 768)
    const float* Wout = (const float*)d_in[4];  // (1024, 1024)
    const float* bo   = (const float*)d_in[5];  // (1024,)
    float* out = (float*)d_out;

    char* ws = (char*)d_ws;
    __bf16* q_bf  = (__bf16*)(ws);                  // 8192*1024*2 = 16 MiB
    __bf16* k_bf  = (__bf16*)(ws + 16777216);       // 2048*1024*2 =  4 MiB
    __bf16* vT_bf = (__bf16*)(ws + 20971520);       // 2048*1024*2 =  4 MiB
    __bf16* a_bf  = (__bf16*)(ws + 25165824);       // 8192*1024*2 = 16 MiB

    ca_proj_q  <<<dim3(128, 8), 256, 0, stream>>>(hs, Wq, q_bf);
    ca_proj_kv <<<dim3(32, 16), 256, 0, stream>>>(ctx, Wkv, k_bf, vT_bf);
    ca_attention<<<dim3(2048), 128, 0, stream>>>(q_bf, k_bf, vT_bf, a_bf);
    ca_proj_out<<<dim3(128, 8), 256, 0, stream>>>(a_bf, Wout, bo, out);
}
